// GaborLayer_40278203302012
// MI455X (gfx1250) — matrix operand order repaired and checked by reading, not compiled, not run on hardware
//
#include <hip/hip_runtime.h>
#include <math.h>
#include <stdint.h>

#define N_GABORS 256
#define H 512
#define W 512
#define HW (H * W)
#define TILES 4
#define TWO_PI 6.28318530717958647692f
#define LOG2E 1.44269504088896340736f

typedef __attribute__((ext_vector_type(16))) _Float16 v16h;
typedef __attribute__((ext_vector_type(2)))  __fp16   v2fp16;  // type returned by cvt_pkrtz
typedef __attribute__((ext_vector_type(8)))  float    v8f;

// ---------------------------------------------------------------------------
// Prep kernel 1: fold per-gabor parameters into 8 floats:
//   [ct, st, offx, g*ct, -g*st, g*offy, kexp, f]
// x_rot          = px*ct + py*st + offx
// gamma * y_rot  = px*(-g*st) + py*(g*ct) + g*offy
// gaussian       = exp2( max(kexp * r2, -80*log2e) ),  kexp = -log2e/(2*sigma^2)
// ---------------------------------------------------------------------------
__global__ void prep_gabor_params(const float* __restrict__ u,
                                  const float* __restrict__ v,
                                  const float* __restrict__ theta,
                                  const float* __restrict__ rel_sigma,
                                  const float* __restrict__ rel_freq,
                                  const float* __restrict__ gma,
                                  float* __restrict__ P) {
  int g = blockIdx.x * blockDim.x + threadIdx.x;
  if (g >= N_GABORS) return;
  float uu = fminf(fmaxf(u[g], -1.f), 1.f);
  float vv = fminf(fmaxf(v[g], -1.f), 1.f);
  float th = fminf(fmaxf(theta[g], 0.f), 1.f) * TWO_PI;
  float ct = cosf(th), st = sinf(th);
  float rs = fminf(fmaxf(rel_sigma[g], -5.f), 5.f);
  float sigma = 16.0f * (0.5f + 0.5f * tanhf(rs));   // base_size = 512/32 = 16
  float gc = fminf(fmaxf(gma[g], -5.f), 5.f);
  float gam = 0.5f + 0.5f / (1.f + expf(-gc));
  float f = expf(rel_freq[g]);
  float kexp = -LOG2E / (2.f * sigma * sigma);
  float* p = P + g * 8;
  p[0] = ct;
  p[1] = st;
  p[2] = -(uu * ct + vv * st);
  p[3] = gam * ct;
  p[4] = -gam * st;
  p[5] = gam * (uu * st - vv * ct);
  p[6] = kexp;
  p[7] = f;
}

// ---------------------------------------------------------------------------
// Prep kernel 2: pack A-matrix coefficients in the dense 16-bit A VGPR layout
// (16x32 f16 per WMMA iteration; K = 2g interleaves cos/sin coefficients):
//   A[m, 2g]   =  0.2*tanh(amp)*cos(2*pi*psi)
//   A[m, 2g+1] = -0.2*tanh(amp)*sin(2*pi*psi)
// Layout entry index: ((iter*32 + lane)*8 + v) -> packed f16x2 (lo = K even).
// Dense A layout: lanes 0-15 (M=lane): V0..3 -> K 0..7, V4..7 -> K 16..23;
//                 lanes 16-31 (M=lane-16): K 8..15 and K 24..31.
// ---------------------------------------------------------------------------
__global__ void prep_acoef(const float* __restrict__ psi,
                           const float* __restrict__ amplitude,
                           uint32_t* __restrict__ AC) {
  int e = blockIdx.x * blockDim.x + threadIdx.x;
  if (e >= 16 * 32 * 8) return;
  int vcol = e & 7;
  int lane = (e >> 3) & 31;
  int iter = e >> 8;
  int m = lane & 15;
  int hf = lane >> 4;
  uint32_t out = 0u;
  if (m < 3) {
    int jA = (vcol & 3) + 4 * hf + 8 * (vcol >> 2);
    int g = iter * 16 + jA;
    float a = fminf(fmaxf(amplitude[g * 3 + m], -5.f), 5.f);
    float amp = 0.2f * tanhf(a);
    float ph = psi[g * 3 + m] * TWO_PI;
    float c1 = amp * cosf(ph);
    float c2 = -amp * sinf(ph);
    union { _Float16 h[2]; uint32_t u; } cv;
    cv.h[0] = (_Float16)c1;   // K even (cos coefficient) in low half
    cv.h[1] = (_Float16)c2;   // K odd  (sin coefficient) in high half
    out = cv.u;
  }
  AC[e] = out;
}

// ---------------------------------------------------------------------------
// Main render kernel: one block per image row, 8 waves x 64 pixels.
// Each wave: 4 N-tiles of 16 pixels, 16 K-iterations of 16 gabors each,
// accumulated with v_wmma_f32_16x16x32_f16.
// B layout (32x16 f16): lanes 0-15 (N=lane): V0..3 -> K 0..7, V4..7 -> K 16..23 (gabors 0..3, 8..11);
//                       lanes 16-31 (N=lane-16): K 8..15 and K 24..31 (gabors 4..7, 12..15).
// ---------------------------------------------------------------------------
__global__ __launch_bounds__(256)
void gabor_render(const float* __restrict__ grid_x,
                  const float* __restrict__ grid_y,
                  const float* __restrict__ P,
                  const uint32_t* __restrict__ AC,
                  float* __restrict__ out) {
  const int y = blockIdx.x;
  const int lane = threadIdx.x & 31;
  const int wave = threadIdx.x >> 5;
  const int hf = lane >> 4;
  const int n = lane & 15;
  const int xbase = wave * 64;

  const float py = grid_y[(size_t)y * W + xbase + n];
  float px[TILES];
#pragma unroll
  for (int t = 0; t < TILES; ++t)
    px[t] = grid_x[(size_t)y * W + xbase + t * 16 + n];

  v8f acc[TILES];
#pragma unroll
  for (int t = 0; t < TILES; ++t) { v8f z = {}; acc[t] = z; }

  for (int iter = 0; iter < 16; ++iter) {
    // A matrix: 8 dwords in this lane's pre-packed layout
    const uint32_t* ap = AC + ((size_t)(iter * 32 + lane)) * 8;
    union { uint4 q[2]; v16h h; } Au;
    Au.q[0] = *(const uint4*)(ap);
    Au.q[1] = *(const uint4*)(ap + 4);

    uint32_t b[TILES][8];
#pragma unroll
    for (int vcol = 0; vcol < 8; ++vcol) {
      const int g = iter * 16 + (vcol & 3) + 4 * hf + 8 * (vcol >> 2);   // B layout gabor mapping
      const float* p = P + (size_t)g * 8;
      float4 p0 = *(const float4*)(p);
      float4 p1 = *(const float4*)(p + 4);
      const float ct = p0.x, st = p0.y, offx = p0.z, gct = p0.w;
      const float gnst = p1.x, goffy = p1.y, kexp = p1.z, f = p1.w;
#pragma unroll
      for (int t = 0; t < TILES; ++t) {
        float xr = fmaf(px[t], ct, fmaf(py, st, offx));
        float yg = fmaf(px[t], gnst, fmaf(py, gct, goffy));
        float r2 = fmaf(yg, yg, xr * xr);
        float ea = fmaxf(kexp * r2, -115.4127f);      // clip(-80) in log2 units
        float e  = __builtin_amdgcn_exp2f(ea);
        float ww = f * xr;                            // revolutions: sin(2*pi*ww)
        float cw = __builtin_amdgcn_cosf(ww);
        float sw = __builtin_amdgcn_sinf(ww);
        union { v2fp16 h; uint32_t u; } pk;
        pk.h = __builtin_amdgcn_cvt_pkrtz(e * cw, e * sw); // lo=cos, hi=sin
        b[t][vcol] = pk.u;
      }
    }
#pragma unroll
    for (int t = 0; t < TILES; ++t) {
      union { uint32_t u[8]; v16h h; } Bu;
#pragma unroll
      for (int vcol = 0; vcol < 8; ++vcol) Bu.u[vcol] = b[t][vcol];
      acc[t] = __builtin_amdgcn_wmma_f32_16x16x32_f16(
          false, Au.h, false, Bu.h, (short)0, acc[t], false, false);
    }
  }

  // C/D layout: lanes 0-15 hold M=0..7 in VGPR 0..7 -> channels 0..2 live in
  // acc[t][0..2] of lanes 0-15.
  if (lane < 16) {
#pragma unroll
    for (int t = 0; t < TILES; ++t) {
      const size_t pix = (size_t)y * W + xbase + t * 16 + n;
#pragma unroll
      for (int c = 0; c < 3; ++c) {
        float r = fminf(fmaxf(acc[t][c], -1.f), 1.f);
        out[(size_t)c * HW + pix] = r;
      }
    }
  }
}

// ---------------------------------------------------------------------------
extern "C" void kernel_launch(void* const* d_in, const int* in_sizes, int n_in,
                              void* d_out, int out_size, void* d_ws, size_t ws_size,
                              hipStream_t stream) {
  const float* grid_x    = (const float*)d_in[0];
  const float* grid_y    = (const float*)d_in[1];
  const float* u         = (const float*)d_in[2];
  const float* v         = (const float*)d_in[3];
  const float* theta     = (const float*)d_in[4];
  const float* rel_sigma = (const float*)d_in[5];
  const float* rel_freq  = (const float*)d_in[6];
  const float* gma       = (const float*)d_in[7];
  const float* psi       = (const float*)d_in[8];
  const float* amplitude = (const float*)d_in[9];

  float*    Pbuf = (float*)d_ws;                                    // 8 KB
  uint32_t* ACbuf = (uint32_t*)((char*)d_ws + N_GABORS * 8 * 4);    // 16 KB

  prep_gabor_params<<<1, 256, 0, stream>>>(u, v, theta, rel_sigma, rel_freq,
                                           gma, Pbuf);
  prep_acoef<<<16, 256, 0, stream>>>(psi, amplitude, ACbuf);
  gabor_render<<<H, 256, 0, stream>>>(grid_x, grid_y, Pbuf, ACbuf,
                                      (float*)d_out);
}
